// StackedInverseFold_34265249087637
// MI455X (gfx1250) — matrix operand order repaired and checked by reading, not compiled, not run on hardware
//
#include <hip/hip_runtime.h>

// ---------------------------------------------------------------------------
// CDNA5 (gfx1250) implementation: bf16 WMMA GEMMs + wave-per-node attention.
// GEMM register-blocks 16x64 per wave: A fragment prepared once per k-chunk,
// reused across 4 column tiles (4 independent WMMA accumulator chains).
// ---------------------------------------------------------------------------

typedef __attribute__((ext_vector_type(16))) __bf16 bf16x16;
typedef __attribute__((ext_vector_type(8)))  float  f32x8;

// ---------------------------------------------------------------------------
// Convert f32 weight [128, ncols] (row-major, in-dim major) to bf16 transposed
// [colPad, 128] (one 256-B contiguous run of K per output column), zero-padded.
// ---------------------------------------------------------------------------
__global__ __launch_bounds__(128)
void convert_wt_kernel(const float* __restrict__ W, __bf16* __restrict__ Wt,
                       int ncols, int colPad)
{
    int idx = blockIdx.x * blockDim.x + threadIdx.x;
    if (idx >= colPad * 128) return;
    int col = idx >> 7;
    int k   = idx & 127;
    float v = (col < ncols) ? W[(size_t)k * ncols + col] : 0.0f;
    Wt[(size_t)col * 128 + k] = (__bf16)v;
}

// ---------------------------------------------------------------------------
// C[M, ncols] = act( (rowScale .* A[M,128]) @ W[128,ncols] + bias )
// One wave per 16x64 output block; K=128 via 4 k-chunks x 4 col-tiles of
// v_wmma_f32_16x16x32_bf16 (16 WMMA per wave).
//
// A fragment (16-bit A 16x32, this code's contiguous k = 16h + s order, matching B): lane = h*16+m,
//   elems 0..7  <- A[row, k0 + 16h     .. k0 + 16h + 7]
//   elems 8..15 <- A[row, k0 + 16h + 8 .. k0 + 16h + 15]
// B fragment (16-bit B 32x16): lane = h*16+n,
//   elems 0..15 <- W[k0+16h .. k0+16h+15, col0+n]  (contiguous in Wt layout)
// C/D: lane h*16+n holds N=col0+n; VGPR r holds M=row0+r+8h.
// ---------------------------------------------------------------------------
__global__ __launch_bounds__(256)
void gemm128_bf16_wmma_kernel(const float* __restrict__ A,
                              const float* __restrict__ rowScale,
                              const __bf16* __restrict__ Wt,
                              const float* __restrict__ bias,
                              float* __restrict__ C,
                              int M, int ncols, int colTiles, int ldC, int doRelu)
{
    const int lane      = threadIdx.x & 31;
    const int waveId    = (blockIdx.x * blockDim.x + threadIdx.x) >> 5;
    const int rowTiles  = (M + 15) >> 4;
    const int colBlocks = (colTiles + 3) >> 2;          // 4 col-tiles per wave
    if (waveId >= rowTiles * colBlocks) return;

    const int rowTile  = waveId / colBlocks;
    const int colBlock = waveId % colBlocks;
    const int row0 = rowTile << 4;

    const int m = lane & 15;
    const int h = lane >> 4;

    int arow = row0 + m;
    if (arow >= M) arow = M - 1;
    const float* __restrict__ Arow = A + (size_t)arow * 128;
    const float s = rowScale ? rowScale[arow] : 1.0f;

    // Per-lane B column pointers for the 4 column tiles (clamped into the
    // zero-padded slot so masked tiles still load safely).
    const __bf16* Wcol[4];
#pragma unroll
    for (int cb = 0; cb < 4; ++cb) {
        int ct = colBlock * 4 + cb;
        if (ct >= colTiles) ct = colTiles - 1;
        Wcol[cb] = Wt + (size_t)((ct << 4) + m) * 128 + 16 * h;
    }

    f32x8 acc[4] = {};

#pragma unroll
    for (int kc = 0; kc < 4; ++kc) {
        const int k0 = kc * 32;

        // ---- A fragment: prepared once, reused by 4 WMMAs ----
        const float4* p0 = (const float4*)(Arow + k0 + 16 * h);
        const float4* p1 = (const float4*)(Arow + k0 + 16 * h + 8);
        float4 x0 = p0[0], x1 = p0[1];
        float4 y0 = p1[0], y1 = p1[1];

        bf16x16 a;
        a[0]  = (__bf16)(s * x0.x); a[1]  = (__bf16)(s * x0.y);
        a[2]  = (__bf16)(s * x0.z); a[3]  = (__bf16)(s * x0.w);
        a[4]  = (__bf16)(s * x1.x); a[5]  = (__bf16)(s * x1.y);
        a[6]  = (__bf16)(s * x1.z); a[7]  = (__bf16)(s * x1.w);
        a[8]  = (__bf16)(s * y0.x); a[9]  = (__bf16)(s * y0.y);
        a[10] = (__bf16)(s * y0.z); a[11] = (__bf16)(s * y0.w);
        a[12] = (__bf16)(s * y1.x); a[13] = (__bf16)(s * y1.y);
        a[14] = (__bf16)(s * y1.z); a[15] = (__bf16)(s * y1.w);

#pragma unroll
        for (int cb = 0; cb < 4; ++cb) {
            bf16x16 b = *(const bf16x16*)(Wcol[cb] + k0);
            acc[cb] = __builtin_amdgcn_wmma_f32_16x16x32_bf16(
                          /*neg_a=*/false, a, /*neg_b=*/false, b,
                          /*c_mod=*/(short)0, acc[cb],
                          /*reuse_a=*/false, /*reuse_b=*/false);
        }
    }

    // ---- epilogue: bias + optional relu, masked stores ----
#pragma unroll
    for (int cb = 0; cb < 4; ++cb) {
        const int colTile = colBlock * 4 + cb;
        if (colTile >= colTiles) break;
        const int col = (colTile << 4) + m;     // lane m doubles as N index
        if (col < ncols) {
            const float bv = bias[col];
#pragma unroll
            for (int r = 0; r < 8; ++r) {
                int row = row0 + r + 8 * h;
                if (row < M) {
                    float v = acc[cb][r] + bv;
                    if (doRelu) v = v > 0.0f ? v : 0.0f;
                    C[(size_t)row * ldC + col] = v;
                }
            }
        }
    }
}

// ---------------------------------------------------------------------------
// out[i] = relu( dot(A[i,0:128], w[0:128]) + b )   (gate MLP 128 -> 1)
// One wave per row; lane covers 4 dims; butterfly reduce.
// ---------------------------------------------------------------------------
__global__ __launch_bounds__(256)
void rowdot_relu_kernel(const float* __restrict__ A, const float* __restrict__ w,
                        const float* __restrict__ b, float* __restrict__ out, int M)
{
    const int node = (blockIdx.x * blockDim.x + threadIdx.x) >> 5;
    const int lane = threadIdx.x & 31;
    if (node >= M) return;
    float4 x = ((const float4*)(A + (size_t)node * 128))[lane];
    float4 y = ((const float4*)w)[lane];
    float s = x.x * y.x + x.y * y.y + x.z * y.z + x.w * y.w;
#pragma unroll
    for (int off = 16; off > 0; off >>= 1) s += __shfl_xor(s, off, 32);
    if (lane == 0) {
        float v = s + b[0];
        out[node] = v > 0.0f ? v : 0.0f;
    }
}

// ---------------------------------------------------------------------------
// Neighbor attention + residual. One wave per node; lane = 4 feature dims.
// kk/vv tables are L2-resident (25.6 MB each << 192 MB L2).
// ---------------------------------------------------------------------------
__global__ __launch_bounds__(128)
void attn_kernel(const float* __restrict__ qq, const float* __restrict__ kk,
                 const float* __restrict__ vv, const int* __restrict__ structure,
                 const float* __restrict__ residual, float* __restrict__ out, int M)
{
    const int node = (blockIdx.x * blockDim.x + threadIdx.x) >> 5;
    const int lane = threadIdx.x & 31;
    if (node >= M) return;

    const float scale = 0.08838834764831845f;  // 1/sqrt(128)
    float4 q = ((const float4*)(qq + (size_t)node * 128))[lane];

    int   nb[16];
    float sc[16];
#pragma unroll
    for (int j = 0; j < 16; ++j) {
        nb[j] = structure[(size_t)node * 16 + j];
        float4 k4 = ((const float4*)(kk + (size_t)nb[j] * 128))[lane];
        float p = q.x * k4.x + q.y * k4.y + q.z * k4.z + q.w * k4.w;
#pragma unroll
        for (int off = 16; off > 0; off >>= 1) p += __shfl_xor(p, off, 32);
        sc[j] = p * scale;
    }

    float mx = sc[0];
#pragma unroll
    for (int j = 1; j < 16; ++j) mx = sc[j] > mx ? sc[j] : mx;
    float sum = 0.0f;
#pragma unroll
    for (int j = 0; j < 16; ++j) { sc[j] = __expf(sc[j] - mx); sum += sc[j]; }
    const float inv = 1.0f / sum;

    float4 acc = {0.0f, 0.0f, 0.0f, 0.0f};
#pragma unroll
    for (int j = 0; j < 16; ++j) {
        float4 v4 = ((const float4*)(vv + (size_t)nb[j] * 128))[lane];
        float w = sc[j];
        acc.x += w * v4.x; acc.y += w * v4.y; acc.z += w * v4.z; acc.w += w * v4.w;
    }

    float4 r = ((const float4*)(residual + (size_t)node * 128))[lane];
    float4 o;
    o.x = acc.x * inv + r.x; o.y = acc.y * inv + r.y;
    o.z = acc.z * inv + r.z; o.w = acc.w * inv + r.w;
    ((float4*)(out + (size_t)node * 128))[lane] = o;
}

// ---------------------------------------------------------------------------
// Host-side orchestration. Leaf order assumes jax sorted-dict flattening:
//  0: inputs [N,128]
//  layer L (base = 1+16L): Wk,Wq,Wv,bk,bq,bv, vW0,vb0,vW1,vb1,vW2,vb2, wW0,wb0,wW1,wb1
//  n_in-7 .. n_in-2: mlp W0,b0,W1,b1,W2,b2
//  n_in-1: structure [N,16] int
// ---------------------------------------------------------------------------
extern "C" void kernel_launch(void* const* d_in, const int* in_sizes, int n_in,
                              void* d_out, int out_size, void* d_ws, size_t ws_size,
                              hipStream_t stream)
{
    const int SIZE = 128, HID = 128, OUTC = 20;
    const int N = in_sizes[0] / SIZE;

    const float* inputs    = (const float*)d_in[0];
    const int*   structure = (const int*)  d_in[n_in - 1];

    const size_t NF = (size_t)N * SIZE;
    float* bufA = (float*)d_ws;
    float* bufB = bufA + NF;
    float* h1   = bufB + NF;
    float* h2   = h1 + NF;
    float* kkb  = h2 + NF;
    float* vvb  = kkb + NF;
    float* wsc  = vvb + NF;                 // [N] gate scalars
    __bf16* warena = (__bf16*)(wsc + N);    // 24 slots of 128x128 bf16
    const size_t SLOT = 128 * 128;
    int slot = 0;
    auto wt = [&](int s_) { return warena + (size_t)s_ * SLOT; };

    auto conv = [&](const void* W, __bf16* dst, int ncols, int colPad) {
        int total = colPad * 128;
        convert_wt_kernel<<<dim3((total + 127) / 128), dim3(128), 0, stream>>>(
            (const float*)W, dst, ncols, colPad);
    };
    auto gemm = [&](const float* A, const float* rs, const __bf16* Wt,
                    const void* bias, float* C, int M, int ncols, int ldC, int relu) {
        int colTiles  = (ncols + 15) / 16;
        int rowTiles  = (M + 15) / 16;
        int colBlocks = (colTiles + 3) / 4;
        long threads = (long)rowTiles * colBlocks * 32;
        int block = 256;
        unsigned grid = (unsigned)((threads + block - 1) / block);
        gemm128_bf16_wmma_kernel<<<dim3(grid), dim3(block), 0, stream>>>(
            A, rs, Wt, (const float*)bias, C, M, ncols, colTiles, ldC, relu);
    };

    const float* cur = inputs;
    float* tgt[2] = {bufA, bufB};

    for (int L = 0; L < 3; ++L) {
        const int base = 1 + 16 * L;
        const void* Wk  = d_in[base + 0];  const void* Wq  = d_in[base + 1];
        const void* Wv  = d_in[base + 2];  const void* bk  = d_in[base + 3];
        const void* bq  = d_in[base + 4];  const void* bv  = d_in[base + 5];
        const void* vW0 = d_in[base + 6];  const void* vb0 = d_in[base + 7];
        const void* vW1 = d_in[base + 8];  const void* vb1 = d_in[base + 9];
        const void* vW2 = d_in[base + 10]; const void* vb2 = d_in[base + 11];
        const void* wW0 = d_in[base + 12]; const void* wb0 = d_in[base + 13];
        const void* wW1 = d_in[base + 14]; const void* wb1 = d_in[base + 15];

        __bf16 *tv0 = wt(slot++), *tv1 = wt(slot++), *tv2 = wt(slot++);
        __bf16 *tw0 = wt(slot++), *tq = wt(slot++), *tk = wt(slot++), *tvv = wt(slot++);
        conv(vW0, tv0, HID, HID);   conv(vW1, tv1, HID, HID);
        conv(vW2, tv2, SIZE, SIZE); conv(wW0, tw0, HID, HID);
        conv(Wq,  tq,  SIZE, SIZE); conv(Wk,  tk,  SIZE, SIZE);
        conv(Wv,  tvv, SIZE, SIZE);

        // value MLP (relu each layer); val ends up in h1
        gemm(cur, nullptr, tv0, vb0, h1, N, HID, HID, 1);
        gemm(h1,  nullptr, tv1, vb1, h2, N, HID, HID, 1);
        gemm(h2,  nullptr, tv2, vb2, h1, N, SIZE, SIZE, 1);
        // gate MLP: hidden in h2, scalar in wsc
        gemm(cur, nullptr, tw0, wb0, h2, N, HID, HID, 1);
        rowdot_relu_kernel<<<dim3((unsigned)(((long)N * 32 + 255) / 256)),
                             dim3(256), 0, stream>>>(h2, (const float*)wW1,
                                                     (const float*)wb1, wsc, N);
        // q/k/v projections of wr = wsc .* val (row-scale fused; gather commutes
        // with the projection, so project all N nodes once, gather in attn)
        gemm(h1, wsc, tq,  bq, h2,  N, SIZE, SIZE, 0);  // qq
        gemm(h1, wsc, tk,  bk, kkb, N, SIZE, SIZE, 0);  // kk
        gemm(h1, wsc, tvv, bv, vvb, N, SIZE, SIZE, 0);  // vv

        float* next = tgt[L & 1];
        attn_kernel<<<dim3((unsigned)(((long)N * 32 + 127) / 128)),
                      dim3(128), 0, stream>>>(h2, kkb, vvb, structure, cur, next, N);
        cur = next;
    }

    // final MLP 128 -> 128 -> 128 -> 20 (relu each, incl. last)
    const void* mW0 = d_in[n_in - 7]; const void* mb0 = d_in[n_in - 6];
    const void* mW1 = d_in[n_in - 5]; const void* mb1 = d_in[n_in - 4];
    const void* mW2 = d_in[n_in - 3]; const void* mb2 = d_in[n_in - 2];
    __bf16 *tm0 = wt(slot++), *tm1 = wt(slot++), *tm2 = wt(slot++);
    conv(mW0, tm0, HID, HID);
    conv(mW1, tm1, HID, HID);
    conv(mW2, tm2, OUTC, 32);

    gemm(cur, nullptr, tm0, mb0, h1, N, HID, HID, 1);
    gemm(h1,  nullptr, tm1, mb1, h2, N, HID, HID, 1);
    gemm(h2,  nullptr, tm2, mb2, (float*)d_out, N, OUTC, OUTC, 1);
    (void)ws_size; (void)out_size;
}
